// LidarCameraFusionMamba_16449724745541
// MI455X (gfx1250) — compile-verified, matrix operand order checked by reading, not run on hardware
//
#include <hip/hip_runtime.h>
#include <hip/hip_bf16.h>

// ---------- types ----------
typedef __attribute__((ext_vector_type(16))) _Float16 v16h;
typedef __attribute__((ext_vector_type(8)))  _Float16 v8h;
typedef __attribute__((ext_vector_type(4)))  _Float16 v4h;
typedef __attribute__((ext_vector_type(8)))  float    v8f;
typedef __attribute__((ext_vector_type(4)))  int      v4i;

// ---------- problem constants ----------
#define BATCHN 4
#define SEQ    1024
#define DM     512
#define DI     1024
#define NSTATE 16
#define MROWS  (BATCHN*SEQ)   // 4096
#define NCH    8
#define LC     (SEQ/NCH)      // 128

// ---------- GEMM tiling ----------
#define TM 64
#define TN 128
#define TK 32
#define LDSA 40   // padded row stride in halves: 80B -> 20-bank lane stride, conflict free, 16B aligned

__device__ __forceinline__ float sigmoidf_(float x){ return 1.f/(1.f+__expf(-x)); }
__device__ __forceinline__ float softplusf_(float x){ return (x>20.f)?x:__logf(1.f+__expf(x)); }
__device__ __forceinline__ float siluf_(float x){ return x*sigmoidf_(x); }

// ---------- async global->LDS copy (gfx1250 ASYNCcnt path), with fallback ----------
#if __has_builtin(__builtin_amdgcn_global_load_async_to_lds_b128)
#define HAVE_ASYNC_LDS 1
typedef __attribute__((address_space(1))) v4i as1_v4i;   // global
typedef __attribute__((address_space(3))) v4i as3_v4i;   // LDS
#else
#define HAVE_ASYNC_LDS 0
#endif

__device__ __forceinline__ void cp16_g2l(const float* g, float* l)
{
#if HAVE_ASYNC_LDS
    __builtin_amdgcn_global_load_async_to_lds_b128(
        (as1_v4i*)(uintptr_t)g,                 // inttoptr -> AS1 v4i*
        (as3_v4i*)(unsigned)(uintptr_t)l,       // low 32 bits of generic addr = LDS offset
        0, 0);
#else
    *(float4*)l = *(const float4*)g;
#endif
}

__device__ __forceinline__ void cp_fence_and_barrier()
{
#if HAVE_ASYNC_LDS
 #if __has_builtin(__builtin_amdgcn_s_wait_asynccnt)
    __builtin_amdgcn_s_wait_asynccnt(0);
 #else
    asm volatile("s_wait_asynccnt 0x0" ::: "memory");
 #endif
#endif
    __syncthreads();
}

// C[M,N] = A[M,K] (row-major fp32, ld lda) * Bw[N,K]^T (Linear weight, ld ldb)
// mode 0: plain store. mode 1: store sigmoid(acc+bias[col]) * aux[row*ldaux+col]
__global__ __launch_bounds__(128)
void gemm_f16wmma(const float* __restrict__ A, int lda,
                  const float* __restrict__ Bw, int ldb,
                  float* __restrict__ C, int ldc,
                  int M, int N, int K,
                  int mode, const float* __restrict__ bias,
                  const float* __restrict__ aux, int ldaux)
{
    __shared__ __align__(16) _Float16 sA[TM*LDSA];
    __shared__ __align__(16) _Float16 sB[TN*LDSA];

    const int tid  = threadIdx.x;
    const int wave = tid >> 5;
    const int lane = tid & 31;
    const int wm   = wave & 1;
    const int wn   = wave >> 1;
    const int m0   = blockIdx.y * TM;
    const int n0   = blockIdx.x * TN;
    const bool fullN = (n0 + TN) <= N;      // block-uniform guard

    v8f acc[2][4];
    {
        v8f z = {0.f,0.f,0.f,0.f,0.f,0.f,0.f,0.f};
        for (int i = 0; i < 2; ++i)
            for (int j = 0; j < 4; ++j) acc[i][j] = z;
    }

    const int lr = tid >> 3;        // 0..15  staging row
    const int lc = (tid & 7) * 4;   // 0,4,...,28 staging col (float4)

    // precomputed row pointers (advance via +k0)
    const float* pA[4];
    const float* pB[8];
    bool bok[8];
    #pragma unroll
    for (int i = 0; i < 4; ++i) pA[i] = A + (size_t)(m0 + i*16 + lr) * lda + lc;
    #pragma unroll
    for (int i = 0; i < 8; ++i) {
        int r = n0 + i*16 + lr;
        bok[i] = r < N;
        pB[i]  = Bw + (size_t)(bok[i] ? r : 0) * ldb + lc;
    }

    float4 ra[4], rb[8];
    // issue first prefetch
    #pragma unroll
    for (int i = 0; i < 4; ++i) ra[i] = *(const float4*)(pA[i]);
    if (fullN) {
        #pragma unroll
        for (int i = 0; i < 8; ++i) rb[i] = *(const float4*)(pB[i]);
    } else {
        #pragma unroll
        for (int i = 0; i < 8; ++i)
            rb[i] = bok[i] ? *(const float4*)(pB[i]) : make_float4(0.f,0.f,0.f,0.f);
    }

    for (int k0 = 0; k0 < K; k0 += TK) {
        // ---- convert prefetched tiles and store to LDS ----
        #pragma unroll
        for (int i = 0; i < 4; ++i) {
            v4h h4; h4[0]=(_Float16)ra[i].x; h4[1]=(_Float16)ra[i].y;
                    h4[2]=(_Float16)ra[i].z; h4[3]=(_Float16)ra[i].w;
            *(v4h*)&sA[(i*16 + lr)*LDSA + lc] = h4;
        }
        #pragma unroll
        for (int i = 0; i < 8; ++i) {
            v4h h4; h4[0]=(_Float16)rb[i].x; h4[1]=(_Float16)rb[i].y;
                    h4[2]=(_Float16)rb[i].z; h4[3]=(_Float16)rb[i].w;
            *(v4h*)&sB[(i*16 + lr)*LDSA + lc] = h4;
        }
        __syncthreads();

        // ---- prefetch next K-step while computing this one ----
        if (k0 + TK < K) {
            const int kn = k0 + TK;
            #pragma unroll
            for (int i = 0; i < 4; ++i) ra[i] = *(const float4*)(pA[i] + kn);
            if (fullN) {
                #pragma unroll
                for (int i = 0; i < 8; ++i) rb[i] = *(const float4*)(pB[i] + kn);
            } else {
                #pragma unroll
                for (int i = 0; i < 8; ++i)
                    rb[i] = bok[i] ? *(const float4*)(pB[i] + kn)
                                   : make_float4(0.f,0.f,0.f,0.f);
            }
        }

        // ---- fragments: two ds_read_b128 each ----
        const int rl    = lane & 15;
        const int khalf = (lane >> 4) * 8;
        v16h af[2], bf[4];
        #pragma unroll
        for (int i = 0; i < 2; ++i) {
            const _Float16* p = &sA[(wm*32 + i*16 + rl)*LDSA + khalf];
            v8h lo = *(const v8h*)p;
            v8h hi = *(const v8h*)(p + 16);
            af[i] = __builtin_shufflevector(lo, hi, 0,1,2,3,4,5,6,7,8,9,10,11,12,13,14,15);
        }
        #pragma unroll
        for (int j = 0; j < 4; ++j) {
            const _Float16* p = &sB[(wn*64 + j*16 + rl)*LDSA + khalf];
            v8h lo = *(const v8h*)p;
            v8h hi = *(const v8h*)(p + 16);
            bf[j] = __builtin_shufflevector(lo, hi, 0,1,2,3,4,5,6,7,8,9,10,11,12,13,14,15);
        }
        #pragma unroll
        for (int i = 0; i < 2; ++i)
            #pragma unroll
            for (int j = 0; j < 4; ++j)
                acc[i][j] = __builtin_amdgcn_wmma_f32_16x16x32_f16(
                    false, af[i], false, bf[j], (short)0, acc[i][j], false, false);
        __syncthreads();
    }

    // ---- store: v8f lane layout: VGPR r -> M = r + 8*(lane>=16), N = lane&15 ----
    const int rbase = (lane >> 4) * 8;
    const int cl    = lane & 15;
    for (int i = 0; i < 2; ++i) {
        for (int j = 0; j < 4; ++j) {
            int col = n0 + wn*64 + j*16 + cl;
            if (col >= N) continue;
            #pragma unroll
            for (int r = 0; r < 8; ++r) {
                int row = m0 + wm*32 + i*16 + rbase + r;
                float v = acc[i][j][r];
                if (mode == 1)
                    v = sigmoidf_(v + bias[col]) * aux[(size_t)row * ldaux + col];
                C[(size_t)row * ldc + col] = v;
            }
        }
    }
}

// ---------- Phase A: per-chunk local scan -> (decay product P, local final h) ----------
__global__ __launch_bounds__(256)
void scan_phaseA(const float* __restrict__ u, const float* __restrict__ dtb,
                 const float* __restrict__ params,
                 const float* __restrict__ Alog_f, const float* __restrict__ Alog_b,
                 const float* __restrict__ bdt,
                 float* __restrict__ cA, float* __restrict__ cH)
{
    __shared__ __align__(16) float sB[LC*NSTATE];
    const int d   = blockIdx.x * 256 + threadIdx.x;
    const int b   = blockIdx.y;
    const int dir = blockIdx.z & 1;
    const int c   = blockIdx.z >> 1;

    // stage B rows (16 floats = 64B each) via async global->LDS, 16B per op
    for (int e = threadIdx.x; e < LC*4; e += 256) {       // 512 x b128 ops
        int t = e >> 2, q = e & 3;
        int l = (dir == 0) ? (c*LC + t) : (SEQ-1 - (c*LC + t));
        cp16_g2l(params + ((size_t)(b*SEQ + l))*64 + 32 + q*4, &sB[t*NSTATE + q*4]);
    }
    cp_fence_and_barrier();

    const float* Alog = (dir == 0) ? Alog_f : Alog_b;
    float Ad[NSTATE];
    #pragma unroll
    for (int n = 0; n < NSTATE; ++n) Ad[n] = -__expf(Alog[d*NSTATE + n]);
    const float db = bdt[d];

    float P[NSTATE], h[NSTATE];
    #pragma unroll
    for (int n = 0; n < NSTATE; ++n) { P[n] = 1.f; h[n] = 0.f; }

    for (int t = 0; t < LC; ++t) {
        int l = (dir == 0) ? (c*LC + t) : (SEQ-1 - (c*LC + t));
        size_t m = (size_t)(b*SEQ + l);
        float dtv = softplusf_(dtb[m*DI + d] + db);
        float du  = dtv * u[m*DI + d];
        #pragma unroll
        for (int n = 0; n < NSTATE; ++n) {
            float dA = __expf(dtv * Ad[n]);
            h[n] = dA * h[n] + du * sB[t*NSTATE + n];
            P[n] *= dA;
        }
    }
    size_t base = (((size_t)dir*4096 + b*DI + d)*NCH + c)*NSTATE;
    #pragma unroll
    for (int n = 0; n < NSTATE; ++n) { cA[base+n] = P[n]; cH[base+n] = h[n]; }
}

// ---------- Phase B: sequential combine across the 8 chunks -> h0 per chunk ----------
__global__ __launch_bounds__(256)
void scan_phaseB(const float* __restrict__ cA, const float* __restrict__ cH,
                 float* __restrict__ h0)
{
    int tid = blockIdx.x * 256 + threadIdx.x;   // 0..131071
    int n   = tid & 15;
    int s   = (tid >> 4) & 4095;
    int dir = tid >> 16;
    size_t base = (((size_t)dir*4096 + s)*NCH)*NSTATE + n;
    float h = 0.f;
    for (int c = 0; c < NCH; ++c) {
        h0[base + c*NSTATE] = h;
        h = cA[base + c*NSTATE] * h + cH[base + c*NSTATE];
    }
}

// ---------- Phase C: replay chunk from h0, emit y (incl. D*u) ----------
__global__ __launch_bounds__(256)
void scan_phaseC(const float* __restrict__ u, const float* __restrict__ dtb,
                 const float* __restrict__ params,
                 const float* __restrict__ Alog_f, const float* __restrict__ Alog_b,
                 const float* __restrict__ bdt, const float* __restrict__ Dp,
                 const float* __restrict__ h0,
                 float* __restrict__ y0, float* __restrict__ y1)
{
    __shared__ __align__(16) float sB[LC*NSTATE];
    __shared__ __align__(16) float sC[LC*NSTATE];
    const int d   = blockIdx.x * 256 + threadIdx.x;
    const int b   = blockIdx.y;
    const int dir = blockIdx.z & 1;
    const int c   = blockIdx.z >> 1;

    for (int e = threadIdx.x; e < LC*4; e += 256) {
        int t = e >> 2, q = e & 3;
        int l = (dir == 0) ? (c*LC + t) : (SEQ-1 - (c*LC + t));
        size_t pm = ((size_t)(b*SEQ + l))*64;
        cp16_g2l(params + pm + 32 + q*4, &sB[t*NSTATE + q*4]);
        cp16_g2l(params + pm + 48 + q*4, &sC[t*NSTATE + q*4]);
    }
    cp_fence_and_barrier();

    const float* Alog = (dir == 0) ? Alog_f : Alog_b;
    float Ad[NSTATE];
    #pragma unroll
    for (int n = 0; n < NSTATE; ++n) Ad[n] = -__expf(Alog[d*NSTATE + n]);
    const float db = bdt[d];
    const float Dd = Dp[d];

    float h[NSTATE];
    size_t hbase = (((size_t)dir*4096 + b*DI + d)*NCH + c)*NSTATE;
    #pragma unroll
    for (int n = 0; n < NSTATE; ++n) h[n] = h0[hbase + n];

    float* yout = (dir == 0) ? y0 : y1;
    for (int t = 0; t < LC; ++t) {
        int l = (dir == 0) ? (c*LC + t) : (SEQ-1 - (c*LC + t));
        size_t m = (size_t)(b*SEQ + l);
        float dtv = softplusf_(dtb[m*DI + d] + db);
        float uv  = u[m*DI + d];
        float du  = dtv * uv;
        float acc = Dd * uv;
        #pragma unroll
        for (int n = 0; n < NSTATE; ++n) {
            float dA = __expf(dtv * Ad[n]);
            h[n] = dA * h[n] + du * sB[t*NSTATE + n];
            acc += sC[t*NSTATE + n] * h[n];
        }
        yout[m*DI + d] = acc;
    }
}

// ---------- combine directions + silu(z) gate ----------
__global__ __launch_bounds__(256)
void combine_gate(const float* __restrict__ y0, const float* __restrict__ y1,
                  const float* __restrict__ camxz, float* __restrict__ yf)
{
    size_t i = (size_t)blockIdx.x * 256 + threadIdx.x;
    size_t m = i >> 10, d = i & 1023;
    float z = camxz[m*2048 + 1024 + d];
    yf[i] = siluf_(z) * (y0[i] + y1[i]);
}

extern "C" void kernel_launch(void* const* d_in, const int* in_sizes, int n_in,
                              void* d_out, int out_size, void* d_ws, size_t ws_size,
                              hipStream_t stream)
{
    const float* lidar  = (const float*)d_in[0];
    const float* camera = (const float*)d_in[1];
    const float* W_in   = (const float*)d_in[2];
    const float* W_x    = (const float*)d_in[3];
    const float* W_dt   = (const float*)d_in[4];
    const float* b_dt   = (const float*)d_in[5];
    const float* Alog_f = (const float*)d_in[6];
    const float* Alog_b = (const float*)d_in[7];
    const float* Dp     = (const float*)d_in[8];
    const float* W_guid = (const float*)d_in[9];
    const float* b_guid = (const float*)d_in[10];
    const float* W_out  = (const float*)d_in[11];
    float* out = (float*)d_out;

    float* ws     = (float*)d_ws;
    float* camxz  = ws;                                     // 4096 x 2048
    float* ubuf   = camxz  + (size_t)MROWS * 2048;          // 4096 x 1024
    float* dtbuf  = ubuf   + (size_t)MROWS * DI;            // 4096 x 1024
    float* params = dtbuf  + (size_t)MROWS * DI;            // 4096 x 64
    float* lidx   = params + (size_t)MROWS * 64;            // 4096 x 1024 (reused as y_fused)
    float* cA     = lidx   + (size_t)MROWS * DI;
    float* cH     = cA     + (size_t)2*4096*NCH*NSTATE;
    float* h0     = cH     + (size_t)2*4096*NCH*NSTATE;
    float* y0     = h0     + (size_t)2*4096*NCH*NSTATE;
    float* y1     = y0     + (size_t)MROWS * DI;

    dim3 blk(128);
    gemm_f16wmma<<<dim3(2048/TN, MROWS/TM), blk, 0, stream>>>(
        camera, DM, W_in, DM, camxz, 2048, MROWS, 2048, DM, 0, nullptr, nullptr, 0);
    gemm_f16wmma<<<dim3(1024/TN, MROWS/TM), blk, 0, stream>>>(
        lidar, DM, W_in, DM, lidx, DI, MROWS, DI, DM, 0, nullptr, nullptr, 0);
    gemm_f16wmma<<<dim3(1024/TN, MROWS/TM), blk, 0, stream>>>(
        lidx, DI, W_guid, DI, ubuf, DI, MROWS, DI, DI, 1, b_guid, camxz, 2048);
    gemm_f16wmma<<<dim3(1, MROWS/TM), blk, 0, stream>>>(
        camxz, 2048, W_x, DI, params, 64, MROWS, 64, DI, 0, nullptr, nullptr, 0);
    gemm_f16wmma<<<dim3(1024/TN, MROWS/TM), blk, 0, stream>>>(
        params, 64, W_dt, 32, dtbuf, DI, MROWS, DI, 32, 0, nullptr, nullptr, 0);

    scan_phaseA<<<dim3(DI/256, BATCHN, NCH*2), dim3(256), 0, stream>>>(
        ubuf, dtbuf, params, Alog_f, Alog_b, b_dt, cA, cH);
    scan_phaseB<<<dim3(512), dim3(256), 0, stream>>>(cA, cH, h0);
    scan_phaseC<<<dim3(DI/256, BATCHN, NCH*2), dim3(256), 0, stream>>>(
        ubuf, dtbuf, params, Alog_f, Alog_b, b_dt, Dp, h0, y0, y1);

    combine_gate<<<dim3((MROWS*DI)/256), dim3(256), 0, stream>>>(y0, y1, camxz, lidx);

    gemm_f16wmma<<<dim3(DM/TN, MROWS/TM), blk, 0, stream>>>(
        lidx, DI, W_out, DI, out, DM, MROWS, DM, DI, 0, nullptr, nullptr, 0);
}
